// MultiHeadAttention1_3375844295036
// MI455X (gfx1250) — matrix operand order repaired and checked by reading, not compiled, not run on hardware
//
#include <hip/hip_runtime.h>
#include <cstdint>
#include <cstddef>

// ---------------------------------------------------------------------------
// Multi-head attention forward (B=2, S=2048, E=1024, H=16, D=64) for gfx1250.
// GEMMs: v_wmma_f32_16x16x32_bf16 with async global->LDS double-buffered B.
// ---------------------------------------------------------------------------

typedef __attribute__((ext_vector_type(16))) __bf16 v16bf;
typedef __attribute__((ext_vector_type(8)))  __bf16 v8bf;
typedef __attribute__((ext_vector_type(8)))  float  v8f;

constexpr int Bc = 2;
constexpr int Sc = 2048;
constexpr int Ec = 1024;
constexpr int Hc = 16;
constexpr int Dc = 64;
constexpr int Mrows = Bc * Sc;          // 4096 token rows
constexpr float SCALE = 0.125f;         // 1/sqrt(64)

#define DI static __device__ __forceinline__

DI v8f wmma_bf16(v16bf a, v16bf b, v8f c) {
  return __builtin_amdgcn_wmma_f32_16x16x32_bf16(false, a, false, b,
                                                 (short)0, c, false, false);
}

// -------- WMMA operand loaders (per ISA 16-bit VGPR layouts) ----------------
// A operand: 16x32 tile of row-major [.. x lda].
DI v16bf load_a16x32(const __bf16* base, int lda, int m0, int k0, int lane) {
  int row = m0 + (lane & 15);
  int kb  = k0 + ((lane & 16) ? 8 : 0);
  const __bf16* p = base + (size_t)row * (size_t)lda + kb;
  v8bf lo = *(const v8bf*)(p);        // K = kb .. kb+7
  v8bf hi = *(const v8bf*)(p + 16);   // K = kb+16 .. kb+23
  v16bf r;
#pragma unroll
  for (int i = 0; i < 8; ++i) { r[i] = lo[i]; r[i + 8] = hi[i]; }
  return r;
}

// B operand: 32x16 tile, where B[k][n] = bt[n][k] (bt row-major [N x K]).
DI v16bf load_bt32x16(const __bf16* bt, int ldb, int n0, int k0, int lane) {
  int row = n0 + (lane & 15);
  int kk  = k0 + ((lane & 16) ? 8 : 0);
  const __bf16* p = bt + (size_t)row * (size_t)ldb + kk;
  v8bf lo = *(const v8bf*)(p);
  v8bf hi = *(const v8bf*)(p + 16);
  v16bf r;
#pragma unroll
  for (int i = 0; i < 8; ++i) { r[i] = lo[i]; r[i + 8] = hi[i]; }
  return r;
}

// Per-lane async DMA of 16 bytes global -> LDS (ASYNCcnt-tracked).
DI void async_b128_to_lds(uint32_t lds_addr, const void* gaddr) {
  asm volatile("global_load_async_to_lds_b128 %0, %1, off"
               :: "v"(lds_addr), "v"((unsigned long long)(size_t)gaddr)
               : "memory");
}
DI void wait_async0() { asm volatile("s_wait_asynccnt 0x0" ::: "memory"); }

// -------- precondition kernels ---------------------------------------------
__global__ __launch_bounds__(256) void cvt_f32_bf16(const float* __restrict__ in,
                                                    __bf16* __restrict__ out, int n) {
  int i = blockIdx.x * blockDim.x + threadIdx.x;
  if (i < n) out[i] = (__bf16)in[i];
}

// out[c][r] = (bf16) in[r][c] ; in is R x C row-major, out is C x R row-major.
__global__ __launch_bounds__(256) void transpose_f32_bf16(const float* __restrict__ in,
                                                          __bf16* __restrict__ out,
                                                          int R, int C) {
  __shared__ float tile[32][33];
  int c0 = blockIdx.x * 32, r0 = blockIdx.y * 32;
#pragma unroll
  for (int i = threadIdx.y; i < 32; i += 8) {
    int r = r0 + i, c = c0 + threadIdx.x;
    tile[i][threadIdx.x] = (r < R && c < C) ? in[(size_t)r * C + c] : 0.0f;
  }
  __syncthreads();
#pragma unroll
  for (int i = threadIdx.y; i < 32; i += 8) {
    int c = c0 + i, r = r0 + threadIdx.x;
    if (c < C && r < R) out[(size_t)c * R + r] = (__bf16)tile[threadIdx.x][i];
  }
}

// ---------------------------------------------------------------------------
// Tiled GEMM core: 256 threads (8 waves), block tile 128(M) x 64(N), K step 32.
// B panel (wT, row-major [N x K]) is double-buffered in LDS via async DMA;
// each wave computes a 16 x 64 strip (4 f32 WMMA accumulators).
// ---------------------------------------------------------------------------
template <int KDIM>
DI void gemm_block_128x64(const __bf16* __restrict__ A,   // [M x KDIM]
                          const __bf16* __restrict__ wT,  // [N x KDIM]
                          __bf16 (*bt_lds)[64 * 32],      // [2][64*32] LDS
                          int m0_blk, int n0_blk,
                          v8f acc[4]) {
  const int tid  = threadIdx.x;
  const int lane = tid & 31;
  const int wv   = tid >> 5;
  const int m0   = m0_blk + wv * 16;

  // async-copy indices: each of the 256 lanes moves one 16B segment
  const int crow = tid >> 2;            // 0..63  (n within tile)
  const int cseg = (tid & 3) * 8;       // 0,8,16,24 (k within tile)

  auto issue = [&](int t, int buf) {
    const __bf16* src = wT + (size_t)(n0_blk + crow) * KDIM + t * 32 + cseg;
    uint32_t dst = (uint32_t)(size_t)&bt_lds[buf][crow * 32 + cseg];
    async_b128_to_lds(dst, src);
  };

  issue(0, 0);                          // prologue
  const int T = KDIM / 32;
  for (int t = 0; t < T; ++t) {
    wait_async0();                      // this wave's DMA into buf[t&1] done
    __syncthreads();                    // all waves' DMA visible; prev compute done
    if (t + 1 < T) issue(t + 1, (t + 1) & 1);

    v16bf a = load_a16x32(A, KDIM, m0, t * 32, lane);
    const __bf16* bl = &bt_lds[t & 1][0];
#pragma unroll
    for (int j = 0; j < 4; ++j)
      acc[j] = wmma_bf16(a, load_bt32x16(bl, 32, 16 * j, 0, lane), acc[j]);
  }
}

// -------- QKV GEMM: qkv = x @ w_attn + b_attn, scattered to Q/K/Vt ---------
__global__ __launch_bounds__(256) void gemm_qkv(const __bf16* __restrict__ xb,
                                                const __bf16* __restrict__ wT,
                                                const float* __restrict__ bias,
                                                __bf16* __restrict__ q_ws,
                                                __bf16* __restrict__ k_ws,
                                                __bf16* __restrict__ vt_ws) {
  __shared__ __align__(16) __bf16 bt_lds[2][64 * 32];
  const int NB = (3 * Ec) / 64;                    // 48 col-blocks
  const int m0_blk = (blockIdx.x / NB) * 128;
  const int n0     = (blockIdx.x % NB) * 64;

  v8f acc[4] = {};
  gemm_block_128x64<Ec>(xb, wT, bt_lds, m0_blk, n0, acc);

  const int lane = threadIdx.x & 31;
  const int wv   = threadIdx.x >> 5;
  const int m0   = m0_blk + wv * 16;
  const int bidx  = m0 / Sc;                       // whole tile in one batch
  const int which = n0 / Ec;                       // 0=Q 1=K 2=V
  const int h     = (n0 % Ec) / Dc;
  const int hi    = (lane >> 4) & 1;
  const int col   = lane & 15;
  const size_t bh = (size_t)(bidx * Hc + h);
#pragma unroll
  for (int j = 0; j < 4; ++j) {
    const int d = 16 * j + col;
    const float bv = bias[n0 + 16 * j + col];
#pragma unroll
    for (int r = 0; r < 8; ++r) {
      const int s = (m0 - bidx * Sc) + r + 8 * hi;
      const float v = acc[j][r] + bv;
      if (which == 0)      q_ws[(bh * Sc + s) * Dc + d] = (__bf16)v;
      else if (which == 1) k_ws[(bh * Sc + s) * Dc + d] = (__bf16)v;
      else                 vt_ws[(bh * Dc + d) * Sc + s] = (__bf16)v;  // V^T
    }
  }
}

// -------- projection GEMM: out = ctx @ w_proj + b_proj (fp32 out) ----------
__global__ __launch_bounds__(256) void gemm_proj(const __bf16* __restrict__ ctx,
                                                 const __bf16* __restrict__ wT,
                                                 const float* __restrict__ bias,
                                                 float* __restrict__ out) {
  __shared__ __align__(16) __bf16 bt_lds[2][64 * 32];
  const int NB = Ec / 64;                          // 16 col-blocks
  const int m0_blk = (blockIdx.x / NB) * 128;
  const int n0     = (blockIdx.x % NB) * 64;

  v8f acc[4] = {};
  gemm_block_128x64<Ec>(ctx, wT, bt_lds, m0_blk, n0, acc);

  const int lane = threadIdx.x & 31;
  const int wv   = threadIdx.x >> 5;
  const int m0   = m0_blk + wv * 16;
  const int hi = (lane >> 4) & 1, col = lane & 15;
#pragma unroll
  for (int j = 0; j < 4; ++j) {
    const float bv = bias[n0 + 16 * j + col];
#pragma unroll
    for (int r = 0; r < 8; ++r)
      out[(size_t)(m0 + r + 8 * hi) * Ec + n0 + 16 * j + col] = acc[j][r] + bv;
  }
}

// -------- causal flash attention: one wave per 16-query tile ---------------
__global__ __launch_bounds__(128) void attn_kernel(const __bf16* __restrict__ q_ws,
                                                   const __bf16* __restrict__ k_ws,
                                                   const __bf16* __restrict__ vt_ws,
                                                   __bf16* __restrict__ ctx_b) {
  __shared__ __align__(16) __bf16 pbuf[4][16 * 32];   // per-wave P staging
  const int lane = threadIdx.x & 31;
  const int wv   = threadIdx.x >> 5;
  const int wid  = blockIdx.x * 4 + wv;
  if (wid >= (Bc * Hc) * (Sc / 16)) return;           // wave-uniform
  const int qt = wid & (Sc / 16 - 1);                 // 0..127
  const int bh = wid >> 7;                            // 0..31
  const int q0 = qt * 16;

  const __bf16* Q  = q_ws  + (size_t)bh * Sc * Dc;
  const __bf16* K  = k_ws  + (size_t)bh * Sc * Dc;
  const __bf16* Vt = vt_ws + (size_t)bh * Dc * Sc;

  const v16bf qa0 = load_a16x32(Q, Dc, q0, 0, lane);
  const v16bf qa1 = load_a16x32(Q, Dc, q0, 32, lane);

  const int hi  = (lane >> 4) & 1;
  const int col = lane & 15;
  float m_i[8], l_i[8];
  v8f acc[4] = {};
#pragma unroll
  for (int r = 0; r < 8; ++r) { m_i[r] = -1e30f; l_i[r] = 0.0f; }

  for (int k0 = 0; k0 < q0 + 16; k0 += 32) {
    v8f s0 = {}, s1 = {};
    s0 = wmma_bf16(qa0, load_bt32x16(K, Dc, k0,      0,  lane), s0);
    s0 = wmma_bf16(qa1, load_bt32x16(K, Dc, k0,      32, lane), s0);
    s1 = wmma_bf16(qa0, load_bt32x16(K, Dc, k0 + 16, 0,  lane), s1);
    s1 = wmma_bf16(qa1, load_bt32x16(K, Dc, k0 + 16, 32, lane), s1);

    float al[8];
#pragma unroll
    for (int r = 0; r < 8; ++r) {
      const int qrow = q0 + r + 8 * hi;
      float a0 = (k0 + col      <= qrow) ? s0[r] * SCALE : -1e30f;
      float a1 = (k0 + 16 + col <= qrow) ? s1[r] * SCALE : -1e30f;
      float mx = fmaxf(a0, a1);
#pragma unroll
      for (int m = 1; m < 16; m <<= 1) mx = fmaxf(mx, __shfl_xor(mx, m, 32));
      const float nm = fmaxf(m_i[r], mx);
      const float p0 = __expf(a0 - nm);
      const float p1 = __expf(a1 - nm);
      float sum = p0 + p1;
#pragma unroll
      for (int m = 1; m < 16; m <<= 1) sum += __shfl_xor(sum, m, 32);
      al[r]  = __expf(m_i[r] - nm);
      l_i[r] = al[r] * l_i[r] + sum;
      m_i[r] = nm;
      const int row = r + 8 * hi;                      // C-layout row
      pbuf[wv][row * 32 + col]      = (__bf16)p0;
      pbuf[wv][row * 32 + 16 + col] = (__bf16)p1;
    }
#pragma unroll
    for (int j = 0; j < 4; ++j)
#pragma unroll
      for (int r = 0; r < 8; ++r) acc[j][r] *= al[r];

    // same-wave LDS ops are in-order; wait + compiler fence before re-reading
    asm volatile("s_wait_dscnt 0" ::: "memory");
    const v16bf pa = load_a16x32(&pbuf[wv][0], 32, 0, 0, lane);
#pragma unroll
    for (int j = 0; j < 4; ++j)
      acc[j] = wmma_bf16(pa, load_bt32x16(Vt, Sc, 16 * j, k0, lane), acc[j]);
    asm volatile("" ::: "memory");
  }

  const int bidx = bh >> 4, h = bh & 15;
#pragma unroll
  for (int j = 0; j < 4; ++j)
#pragma unroll
    for (int r = 0; r < 8; ++r) {
      const int srow = q0 + r + 8 * hi;
      const float v  = acc[j][r] / fmaxf(l_i[r], 1e-20f);
      ctx_b[((size_t)(bidx * Sc + srow)) * Ec + h * Dc + 16 * j + col] = (__bf16)v;
    }
}

// ---------------------------------------------------------------------------
extern "C" void kernel_launch(void* const* d_in, const int* in_sizes, int n_in,
                              void* d_out, int out_size, void* d_ws, size_t ws_size,
                              hipStream_t stream) {
  (void)in_sizes; (void)n_in; (void)out_size; (void)ws_size;
  const float* x      = (const float*)d_in[0];
  const float* w_attn = (const float*)d_in[1];
  const float* b_attn = (const float*)d_in[2];
  const float* w_proj = (const float*)d_in[3];
  const float* b_proj = (const float*)d_in[4];
  float* out = (float*)d_out;

  char* ws = (char*)d_ws;
  size_t off = 0;
  auto take = [&](size_t elems) { __bf16* p = (__bf16*)(ws + off); off += elems * 2; return p; };
  __bf16* xb   = take((size_t)Mrows * Ec);          // 8 MB
  __bf16* wTa  = take((size_t)3 * Ec * Ec);         // 6 MB
  __bf16* wTp  = take((size_t)Ec * Ec);             // 2 MB
  __bf16* q_ws = take((size_t)Bc * Hc * Sc * Dc);   // 8 MB
  __bf16* k_ws = take((size_t)Bc * Hc * Sc * Dc);   // 8 MB
  __bf16* vt_ws= take((size_t)Bc * Hc * Dc * Sc);   // 8 MB
  __bf16* ctxb = take((size_t)Mrows * Ec);          // 8 MB  (48 MB total)

  // 1. convert / transpose inputs to bf16
  {
    int n = Mrows * Ec;
    cvt_f32_bf16<<<(n + 255) / 256, 256, 0, stream>>>(x, xb, n);
  }
  transpose_f32_bf16<<<dim3(3 * Ec / 32, Ec / 32), dim3(32, 8), 0, stream>>>(w_attn, wTa, Ec, 3 * Ec);
  transpose_f32_bf16<<<dim3(Ec / 32, Ec / 32),     dim3(32, 8), 0, stream>>>(w_proj, wTp, Ec, Ec);

  // 2. QKV GEMM: (4096/128) x (3072/64) = 32*48 = 1536 block tiles
  gemm_qkv<<<1536, 256, 0, stream>>>(xb, wTa, b_attn, q_ws, k_ws, vt_ws);

  // 3. causal attention: 32 (b,h) * 128 q-tiles = 4096 waves, 4 per block
  attn_kernel<<<4096 / 4, 128, 0, stream>>>(q_ws, k_ws, vt_ws, ctxb);

  // 4. projection: (4096/128) x (1024/64) = 32*16 = 512 block tiles
  gemm_proj<<<512, 256, 0, stream>>>(ctxb, wTp, b_proj, out);
}
